// TransolverBlock_30709016167113
// MI455X (gfx1250) — matrix operand order repaired and checked by reading, not compiled, not run on hardware
//
#include <hip/hip_runtime.h>
#include <hip/hip_bf16.h>
#include <cstdint>

#define BB 4
#define NNTOK 8192
#define DIMC 512
#define HH 8
#define DHH 64
#define KSL 64
#define INNERC 512
#define HIDC 1024

typedef __bf16 bf16;
typedef __attribute__((ext_vector_type(16))) __bf16 v16bf;
typedef __attribute__((ext_vector_type(8)))  float   v8f;
typedef __attribute__((ext_vector_type(4)))  int     i32x4;
typedef __attribute__((ext_vector_type(8)))  int     i32x8;

union FragU { v16bf v; unsigned u[8]; };

__device__ __forceinline__ v8f wmma_bf(v16bf a, v16bf b, v8f c) {
  return __builtin_amdgcn_wmma_f32_16x16x32_bf16(false, a, false, b, (short)0, c,
                                                 false, false);
}

__device__ __forceinline__ v8f zero8() {
  v8f z;
#pragma unroll
  for (int i = 0; i < 8; ++i) z[i] = 0.0f;
  return z;
}

__device__ __forceinline__ float gelu_f(float x) {
  return 0.5f * x * (1.0f + erff(x * 0.70710678118654752f));
}

// Async global -> LDS copy of 16 bytes per lane (ASYNCcnt tracked).
__device__ __forceinline__ void async_copy_b128(const void* gsrc, void* ldst) {
  unsigned lds_off = (unsigned)(uintptr_t)ldst;          // LDS aperture: low 32 = offset
  unsigned long long ga = (unsigned long long)(uintptr_t)gsrc;
  asm volatile("global_load_async_to_lds_b128 %0, %1, off"
               :
               : "v"(lds_off), "v"(ga)
               : "memory");
}

__device__ __forceinline__ void async_wait0() {
  asm volatile("s_wait_asynccnt 0x0" ::: "memory");
}

// TDM: load a 2D tile rows x 64 bf16 (row stride 64 elems in memory) into LDS,
// padding 16B after each 128B row => LDS row stride 72 bf16 elements.
__device__ __forceinline__ void tdm_load_tile_128x64(const void* gsrc,
                                                     void* ldst, int rows) {
  unsigned lds_off = (unsigned)(uintptr_t)ldst;
  unsigned long long ga = (unsigned long long)(uintptr_t)gsrc;
  i32x4 g0;
  g0[0] = 1;                                          // count=1, user descriptor
  g0[1] = (int)lds_off;                               // lds_addr
  g0[2] = (int)(unsigned)(ga & 0xffffffffull);        // global_addr[31:0]
  g0[3] = (int)(((unsigned)(ga >> 32) & 0x01ffffffu)  // global_addr[56:32]
                | (2u << 30));                        // type = 2 ("image")
  i32x8 g1;
  g1[0] = (1 << 16)      // data_size = 2 bytes
          | (1 << 20)    // pad_enable
          | (4 << 22)    // pad_interval: 32 DWORDs (128B) between pads
          | (3 << 25);   // pad_amount: 4 DWORDs (16B)
  g1[1] = (64 << 16);            // tensor_dim0 = 64 (bits 79:48 low half)
  g1[2] = (rows << 16);          // tensor_dim0 hi=0 | tensor_dim1 low16 = rows
  g1[3] = (64 << 16);            // tensor_dim1 hi=0 | tile_dim0 = 64
  g1[4] = rows;                  // tile_dim1 = rows, tile_dim2 = 0
  g1[5] = 64;                    // tensor_dim0_stride low32 = 64 elems
  g1[6] = 0;                     // stride hi | tensor_dim1_stride low
  g1[7] = 0;
  asm volatile("tensor_load_to_lds %0, %1" : : "s"(g0), "s"(g1) : "memory");
  __builtin_amdgcn_s_wait_tensorcnt(0);
}

// A fragment: 16x32 bf16 tile, row m, K-contiguous. base -> element (m0, k0).
__device__ __forceinline__ v16bf frag_a(const bf16* base, int stride) {
  int lane = threadIdx.x & 31;
  int m = lane & 15, hi = lane >> 4;
  const bf16* row = base + m * stride + hi * 8;
  FragU f;
  f.u[0] = *(const unsigned*)(row + 0);
  f.u[1] = *(const unsigned*)(row + 2);
  f.u[2] = *(const unsigned*)(row + 4);
  f.u[3] = *(const unsigned*)(row + 6);
  f.u[4] = *(const unsigned*)(row + 16);
  f.u[5] = *(const unsigned*)(row + 18);
  f.u[6] = *(const unsigned*)(row + 20);
  f.u[7] = *(const unsigned*)(row + 22);
  return f.v;
}

// B fragment: 32x16 bf16 from Bt storage (row n, K-contiguous).
__device__ __forceinline__ v16bf frag_b(const bf16* base, int stride) {
  int lane = threadIdx.x & 31;
  int n = lane & 15, hi = lane >> 4;
  const bf16* row = base + n * stride + hi * 8;
  FragU f;
#pragma unroll
  for (int j = 0; j < 8; ++j) f.u[j] = *(const unsigned*)(row + 2 * j + (j >= 4 ? 8 : 0));
  return f.v;
}

// ---------------------------------------------------------------------------
// Generic tiled GEMM: C[M,N] = A[M,K](bf16) * Bt[N,K]^T (bf16) + bias, epilogue
// EPI: 0=bias, 1=bias+gelu, 2=bias+residual(fp32)
// Tiles staged via GLOBAL_LOAD_ASYNC_TO_LDS_B128; next stage prefetched.
// ---------------------------------------------------------------------------
template <typename OUT, int EPI>
__global__ __launch_bounds__(256) void gemm_bf16(
    const bf16* __restrict__ A, const bf16* __restrict__ Bt,
    const float* __restrict__ bias, const float* __restrict__ res,
    OUT* __restrict__ C, int M, int N, int K) {
  __shared__ bf16 As[128][72];
  __shared__ bf16 Bs[128][72];
  int tid = threadIdx.x;
  int lane = tid & 31, wave = tid >> 5;
  int wm = wave & 1, wn = wave >> 1;
  int row0 = blockIdx.y * 128, col0 = blockIdx.x * 128;

  v8f acc[4][2];
#pragma unroll
  for (int mi = 0; mi < 4; ++mi)
#pragma unroll
    for (int ni = 0; ni < 2; ++ni) acc[mi][ni] = zero8();

  for (int k0 = 0; k0 < K; k0 += 64) {
#pragma unroll
    for (int i = 0; i < 4; ++i) {
      int e = tid + i * 256;
      int r = e >> 3, s = e & 7;
      async_copy_b128(A + (size_t)(row0 + r) * K + k0 + s * 8, &As[r][s * 8]);
      async_copy_b128(Bt + (size_t)(col0 + r) * K + k0 + s * 8, &Bs[r][s * 8]);
      if (k0 + 64 < K) {  // warm next K-stage into L2 (global_prefetch_b8)
        __builtin_prefetch((const void*)(A + (size_t)(row0 + r) * K + k0 + 64 + s * 8), 0, 1);
        __builtin_prefetch((const void*)(Bt + (size_t)(col0 + r) * K + k0 + 64 + s * 8), 0, 1);
      }
    }
    async_wait0();
    __syncthreads();
#pragma unroll
    for (int kk = 0; kk < 64; kk += 32) {
      v16bf af[4], bfr[2];
#pragma unroll
      for (int mi = 0; mi < 4; ++mi)
        af[mi] = frag_a(&As[wm * 64 + mi * 16][kk], 72);
#pragma unroll
      for (int ni = 0; ni < 2; ++ni)
        bfr[ni] = frag_b(&Bs[wn * 32 + ni * 16][kk], 72);
#pragma unroll
      for (int mi = 0; mi < 4; ++mi)
#pragma unroll
        for (int ni = 0; ni < 2; ++ni)
          acc[mi][ni] = wmma_bf(af[mi], bfr[ni], acc[mi][ni]);
    }
    __syncthreads();
  }

  int hi = lane >> 4, nl = lane & 15;
#pragma unroll
  for (int mi = 0; mi < 4; ++mi)
#pragma unroll
    for (int ni = 0; ni < 2; ++ni)
#pragma unroll
      for (int r = 0; r < 8; ++r) {
        int row = row0 + wm * 64 + mi * 16 + r + 8 * hi;
        int col = col0 + wn * 32 + ni * 16 + nl;
        float v = acc[mi][ni][r] + bias[col];
        if constexpr (EPI == 1) v = gelu_f(v);
        if constexpr (EPI == 2) v += res[(size_t)row * N + col];
        C[(size_t)row * N + col] = (OUT)v;
      }
}

// ---------------------------------------------------------------------------
// Conversion kernels
// ---------------------------------------------------------------------------
__global__ void f2bf(const float* __restrict__ in, bf16* __restrict__ out, int n) {
  int i = blockIdx.x * 256 + threadIdx.x;
  if (i < n) out[i] = (bf16)in[i];
}

__global__ void trans_f2bf(const float* __restrict__ W, bf16* __restrict__ Wt,
                           int K, int N) {
  int i = blockIdx.x * 256 + threadIdx.x;
  if (i < K * N) {
    int k = i / N, n = i - k * N;
    Wt[(size_t)n * K + k] = (bf16)W[i];
  }
}

// ---------------------------------------------------------------------------
// Per-token kernel: temp MLP + logits + gumbel softmax -> sw (bf16)
// grid (N/128, B*H), 256 threads
// ---------------------------------------------------------------------------
__global__ __launch_bounds__(256) void token_kernel(
    const bf16* __restrict__ xm, const float* __restrict__ gumbel,
    const bf16* __restrict__ WslT, const float* __restrict__ bsl,
    const bf16* __restrict__ tw1T, const float* __restrict__ tb1,
    const float* __restrict__ tw2, const float* __restrict__ tb2,
    const float* __restrict__ hbias, bf16* __restrict__ swout) {
  __shared__ float fbuf[128][64];
  __shared__ float trow[128];
  int bh = blockIdx.y, b = bh >> 3, h = bh & 7;
  int n0 = blockIdx.x * 128;
  int tid = threadIdx.x, lane = tid & 31, wave = tid >> 5;
  int hi = lane >> 4, nl = lane & 15;

  const bf16* abase =
      xm + ((size_t)(b * NNTOK + n0 + wave * 16)) * INNERC + h * DHH;
  v16bf af[2];
  af[0] = frag_a(abase + 0, INNERC);
  af[1] = frag_a(abase + 32, INNERC);

  // t1 = xm @ tw1
  v8f acc[4];
#pragma unroll
  for (int t = 0; t < 4; ++t) acc[t] = zero8();
#pragma unroll
  for (int kk = 0; kk < 2; ++kk)
#pragma unroll
    for (int t = 0; t < 4; ++t)
      acc[t] = wmma_bf(af[kk], frag_b(tw1T + (t * 16) * 64 + kk * 32, 64), acc[t]);
#pragma unroll
  for (int t = 0; t < 4; ++t)
#pragma unroll
    for (int r = 0; r < 8; ++r) {
      int row = wave * 16 + r + 8 * hi;
      int col = t * 16 + nl;
      fbuf[row][col] = gelu_f(acc[t][r] + tb1[col]) * tw2[col];
    }
  __syncthreads();
  {  // temp per row
    int row = tid >> 1, hf = tid & 1;
    float s = 0.0f;
#pragma unroll
    for (int j = 0; j < 32; ++j) s += fbuf[row][hf * 32 + j];
    s += __shfl_xor(s, 1, 32);
    if (hf == 0) {
      float tv = gelu_f(s + tb2[0]) + hbias[h];
      trow[row] = fmaxf(tv, 0.01f);
    }
  }
  __syncthreads();

  // logits = xm @ Wsl
#pragma unroll
  for (int t = 0; t < 4; ++t) acc[t] = zero8();
#pragma unroll
  for (int kk = 0; kk < 2; ++kk)
#pragma unroll
    for (int t = 0; t < 4; ++t)
      acc[t] = wmma_bf(af[kk], frag_b(WslT + (t * 16) * 64 + kk * 32, 64), acc[t]);
#pragma unroll
  for (int t = 0; t < 4; ++t)
#pragma unroll
    for (int r = 0; r < 8; ++r) {
      int row = wave * 16 + r + 8 * hi;
      int col = t * 16 + nl;
      fbuf[row][col] = acc[t][r] + bsl[col];
    }
  __syncthreads();
  {  // gumbel softmax per row (2 threads/row, 32 cols each)
    int row = tid >> 1, hf = tid & 1;
    const float* grow = gumbel + ((size_t)bh * NNTOK + n0 + row) * KSL;
    float tinv = 1.0f / trow[row];
    float m = -1e30f;
#pragma unroll
    for (int j = 0; j < 32; ++j) {
      int c = hf * 32 + j;
      float l = (fbuf[row][c] + grow[c]) * tinv;
      fbuf[row][c] = l;
      m = fmaxf(m, l);
    }
    m = fmaxf(m, __shfl_xor(m, 1, 32));
    float s = 0.0f;
#pragma unroll
    for (int j = 0; j < 32; ++j) s += __expf(fbuf[row][hf * 32 + j] - m);
    s += __shfl_xor(s, 1, 32);
    float inv = 1.0f / s;
    bf16* orow = swout + ((size_t)bh * NNTOK + n0 + row) * KSL;
#pragma unroll
    for (int j = 0; j < 32; ++j) {
      int c = hf * 32 + j;
      orow[c] = (bf16)(__expf(fbuf[row][c] - m) * inv);
    }
  }
}

// ---------------------------------------------------------------------------
// Dispatch: st[k,c] = (sum_n sw[n,k]*xm[n,c]) / (snorm[k]+1e-5); one block/(b,h)
// ---------------------------------------------------------------------------
__global__ __launch_bounds__(256) void dispatch_kernel(
    const bf16* __restrict__ sw, const bf16* __restrict__ xm,
    bf16* __restrict__ st) {
  __shared__ bf16 swT[64][66];
  __shared__ bf16 xmT[64][66];
  __shared__ float snr[64][4];
  __shared__ float sn[64];
  int bh = blockIdx.x, b = bh >> 3, h = bh & 7;
  int tid = threadIdx.x, lane = tid & 31, wave = tid >> 5;
  int mi = wave & 3, ni = wave >> 2;
  int kcol = tid & 63, q = tid >> 6;

  v8f acc[2];
  acc[0] = zero8();
  acc[1] = zero8();
  float snp = 0.0f;

  for (int n0 = 0; n0 < NNTOK; n0 += 64) {
#pragma unroll
    for (int i = 0; i < 16; ++i) {
      int e = tid + i * 256;
      int n = e >> 6, c = e & 63;
      swT[c][n] = sw[((size_t)bh * NNTOK + n0 + n) * KSL + c];
      xmT[c][n] = xm[((size_t)(b * NNTOK + n0 + n)) * INNERC + h * DHH + c];
    }
    __syncthreads();
#pragma unroll
    for (int j = 0; j < 16; ++j) snp += (float)swT[kcol][q * 16 + j];
#pragma unroll
    for (int kk = 0; kk < 64; kk += 32) {
      v16bf a = frag_a(&swT[mi * 16][kk], 66);
#pragma unroll
      for (int t = 0; t < 2; ++t)
        acc[t] = wmma_bf(a, frag_b(&xmT[ni * 32 + t * 16][kk], 66), acc[t]);
    }
    __syncthreads();
  }
  snr[kcol][q] = snp;
  __syncthreads();
  if (tid < 64)
    sn[tid] = snr[tid][0] + snr[tid][1] + snr[tid][2] + snr[tid][3] + 1e-5f;
  __syncthreads();

  int hi = lane >> 4, nl = lane & 15;
#pragma unroll
  for (int t = 0; t < 2; ++t)
#pragma unroll
    for (int r = 0; r < 8; ++r) {
      int krow = mi * 16 + r + 8 * hi;
      int c = ni * 32 + t * 16 + nl;
      st[(size_t)bh * 4096 + krow * 64 + c] = (bf16)(acc[t][r] / sn[krow]);
    }
}

// ---------------------------------------------------------------------------
// Attention over slice tokens, one block/(b,h). Emits out_st^T (c-major).
// ---------------------------------------------------------------------------
__global__ __launch_bounds__(256) void attn_kernel(
    const bf16* __restrict__ st, const bf16* __restrict__ WqT,
    const bf16* __restrict__ WkT, const bf16* __restrict__ WvT,
    bf16* __restrict__ outT) {
  __shared__ bf16 qs[64][66], ks[64][66], vT[64][66];
  __shared__ float sc[64][64];
  __shared__ bf16 at[64][66];
  int bh = blockIdx.x;
  int tid = threadIdx.x, lane = tid & 31, wave = tid >> 5;
  int mi = wave & 3, ni = wave >> 2;
  int hi = lane >> 4, nl = lane & 15;
  const bf16* stb = st + (size_t)bh * 4096;

  v8f aq[2], ak[2], av[2];
#pragma unroll
  for (int t = 0; t < 2; ++t) { aq[t] = zero8(); ak[t] = zero8(); av[t] = zero8(); }
#pragma unroll
  for (int kk = 0; kk < 64; kk += 32) {
    v16bf a = frag_a(stb + (mi * 16) * 64 + kk, 64);
#pragma unroll
    for (int t = 0; t < 2; ++t) {
      int cb = (ni * 32 + t * 16) * 64 + kk;
      aq[t] = wmma_bf(a, frag_b(WqT + cb, 64), aq[t]);
      ak[t] = wmma_bf(a, frag_b(WkT + cb, 64), ak[t]);
      av[t] = wmma_bf(a, frag_b(WvT + cb, 64), av[t]);
    }
  }
#pragma unroll
  for (int t = 0; t < 2; ++t)
#pragma unroll
    for (int r = 0; r < 8; ++r) {
      int row = mi * 16 + r + 8 * hi;
      int col = ni * 32 + t * 16 + nl;
      qs[row][col] = (bf16)aq[t][r];
      ks[row][col] = (bf16)ak[t][r];
      vT[col][row] = (bf16)av[t][r];
    }
  __syncthreads();

  v8f as2[2];
  as2[0] = zero8();
  as2[1] = zero8();
#pragma unroll
  for (int kk = 0; kk < 64; kk += 32) {
    v16bf a = frag_a(&qs[mi * 16][kk], 66);
#pragma unroll
    for (int t = 0; t < 2; ++t)
      as2[t] = wmma_bf(a, frag_b(&ks[ni * 32 + t * 16][kk], 66), as2[t]);
  }
#pragma unroll
  for (int t = 0; t < 2; ++t)
#pragma unroll
    for (int r = 0; r < 8; ++r) {
      int row = mi * 16 + r + 8 * hi;
      int col = ni * 32 + t * 16 + nl;
      sc[row][col] = as2[t][r] * 0.125f;
    }
  __syncthreads();
  {  // softmax: 4 threads/row, 16 cols each
    int row = tid >> 2, qd = tid & 3;
    float m = -1e30f;
#pragma unroll
    for (int j = 0; j < 16; ++j) m = fmaxf(m, sc[row][qd * 16 + j]);
    m = fmaxf(m, __shfl_xor(m, 1, 32));
    m = fmaxf(m, __shfl_xor(m, 2, 32));
    float s = 0.0f;
#pragma unroll
    for (int j = 0; j < 16; ++j) s += __expf(sc[row][qd * 16 + j] - m);
    s += __shfl_xor(s, 1, 32);
    s += __shfl_xor(s, 2, 32);
    float inv = 1.0f / s;
#pragma unroll
    for (int j = 0; j < 16; ++j)
      at[row][qd * 16 + j] = (bf16)(__expf(sc[row][qd * 16 + j] - m) * inv);
  }
  __syncthreads();

  v8f ao[2];
  ao[0] = zero8();
  ao[1] = zero8();
#pragma unroll
  for (int kk = 0; kk < 64; kk += 32) {
    v16bf a = frag_a(&at[mi * 16][kk], 66);
#pragma unroll
    for (int t = 0; t < 2; ++t)
      ao[t] = wmma_bf(a, frag_b(&vT[ni * 32 + t * 16][kk], 66), ao[t]);
  }
#pragma unroll
  for (int t = 0; t < 2; ++t)
#pragma unroll
    for (int r = 0; r < 8; ++r) {
      int row = mi * 16 + r + 8 * hi;            // slice token g
      int col = ni * 32 + t * 16 + nl;           // channel c
      outT[(size_t)bh * 4096 + col * 64 + row] = (bf16)ao[t][r];
    }
}

// ---------------------------------------------------------------------------
// Scatter back: y[n, h*64+c] = sum_g sw[n,g] * out_st[g,c]; grid (N/128, B*H)
// sw tile pulled by the Tensor Data Mover into padded LDS (stride 72).
// ---------------------------------------------------------------------------
__global__ __launch_bounds__(256) void scatter_kernel(
    const bf16* __restrict__ sw, const bf16* __restrict__ outT,
    bf16* __restrict__ y) {
  __shared__ bf16 swA[128][72];
  int bh = blockIdx.y, b = bh >> 3, h = bh & 7;
  int n0 = blockIdx.x * 128;
  int tid = threadIdx.x, lane = tid & 31, wave = tid >> 5;
  if (wave == 0) {
    tdm_load_tile_128x64(sw + ((size_t)bh * NNTOK + n0) * KSL, &swA[0][0], 128);
  }
  __syncthreads();
  const bf16* Btg = outT + (size_t)bh * 4096;
  v8f acc[4];
#pragma unroll
  for (int t = 0; t < 4; ++t) acc[t] = zero8();
#pragma unroll
  for (int kk = 0; kk < 64; kk += 32) {
    v16bf a = frag_a(&swA[wave * 16][kk], 72);
#pragma unroll
    for (int t = 0; t < 4; ++t)
      acc[t] = wmma_bf(a, frag_b(Btg + (t * 16) * 64 + kk, 64), acc[t]);
  }
  int hi = lane >> 4, nl = lane & 15;
#pragma unroll
  for (int t = 0; t < 4; ++t)
#pragma unroll
    for (int r = 0; r < 8; ++r) {
      int row = n0 + wave * 16 + r + 8 * hi;
      int c = t * 16 + nl;
      y[((size_t)(b * NNTOK + row)) * INNERC + h * DHH + c] = (bf16)acc[t][r];
    }
}

// ---------------------------------------------------------------------------
// LayerNorm over 512, one block per row
// ---------------------------------------------------------------------------
__global__ __launch_bounds__(512) void ln_kernel(const float* __restrict__ in,
                                                 const float* __restrict__ g,
                                                 const float* __restrict__ bta,
                                                 bf16* __restrict__ out) {
  __shared__ float red[512];
  int row = blockIdx.x, tid = threadIdx.x;
  float v = in[(size_t)row * DIMC + tid];
  red[tid] = v;
  __syncthreads();
  for (int s = 256; s > 0; s >>= 1) {
    if (tid < s) red[tid] += red[tid + s];
    __syncthreads();
  }
  float mu = red[0] * (1.0f / DIMC);
  __syncthreads();
  float d = v - mu;
  red[tid] = d * d;
  __syncthreads();
  for (int s = 256; s > 0; s >>= 1) {
    if (tid < s) red[tid] += red[tid + s];
    __syncthreads();
  }
  float var = red[0] * (1.0f / DIMC);
  float rs = rsqrtf(var + 1e-5f);
  out[(size_t)row * DIMC + tid] = (bf16)(g[tid] * d * rs + bta[tid]);
}

// ---------------------------------------------------------------------------
extern "C" void kernel_launch(void* const* d_in, const int* in_sizes, int n_in,
                              void* d_out, int out_size, void* d_ws,
                              size_t ws_size, hipStream_t stream) {
  const float* x      = (const float*)d_in[0];
  const float* gumbel = (const float*)d_in[1];
  const float* Wx     = (const float*)d_in[2];
  const float* bx     = (const float*)d_in[3];
  const float* tw1    = (const float*)d_in[4];
  const float* tb1    = (const float*)d_in[5];
  const float* tw2    = (const float*)d_in[6];
  const float* tb2    = (const float*)d_in[7];
  const float* hbias  = (const float*)d_in[8];
  const float* Wsl    = (const float*)d_in[9];
  const float* bsl    = (const float*)d_in[10];
  const float* Wq     = (const float*)d_in[11];
  const float* Wk     = (const float*)d_in[12];
  const float* Wv     = (const float*)d_in[13];
  const float* Wp     = (const float*)d_in[14];
  const float* bp     = (const float*)d_in[15];
  const float* ln2g   = (const float*)d_in[16];
  const float* ln2b   = (const float*)d_in[17];
  const float* W1     = (const float*)d_in[18];
  const float* b1     = (const float*)d_in[19];
  const float* W2     = (const float*)d_in[20];
  const float* b2     = (const float*)d_in[21];

  char* ws = (char*)d_ws;
  size_t off = 0;
  auto take = [&](size_t bytes) -> char* {
    char* p = ws + off;
    off += (bytes + 255) & ~size_t(255);
    return p;
  };
  const size_t MT = (size_t)BB * NNTOK;  // 32768 tokens
  bf16* x_bf  = (bf16*)take(MT * DIMC * 2);
  bf16* xm_bf = (bf16*)take(MT * INNERC * 2);
  bf16* sw_bf = (bf16*)take((size_t)BB * HH * NNTOK * KSL * 2);
  bf16* st_bf = (bf16*)take((size_t)BB * HH * KSL * DHH * 2);
  bf16* ostT  = (bf16*)take((size_t)BB * HH * KSL * DHH * 2);
  bf16* y_bf  = (bf16*)take(MT * INNERC * 2);
  float* out2 = (float*)take(MT * DIMC * 4);
  bf16* ln_bf = (bf16*)take(MT * DIMC * 2);
  bf16* h1_bf = (bf16*)take(MT * HIDC * 2);
  bf16* WxT  = (bf16*)take(512 * 512 * 2);
  bf16* tw1T = (bf16*)take(64 * 64 * 2);
  bf16* WslT = (bf16*)take(64 * 64 * 2);
  bf16* WqT  = (bf16*)take(64 * 64 * 2);
  bf16* WkT  = (bf16*)take(64 * 64 * 2);
  bf16* WvT  = (bf16*)take(64 * 64 * 2);
  bf16* WpT  = (bf16*)take(512 * 512 * 2);
  bf16* W1T  = (bf16*)take(512 * 1024 * 2);
  bf16* W2T  = (bf16*)take(1024 * 512 * 2);

  dim3 blk(256);
  int nx = (int)(MT * DIMC);
  f2bf<<<(nx + 255) / 256, blk, 0, stream>>>(x, x_bf, nx);
  trans_f2bf<<<(512 * 512 + 255) / 256, blk, 0, stream>>>(Wx, WxT, 512, 512);
  trans_f2bf<<<(64 * 64 + 255) / 256, blk, 0, stream>>>(tw1, tw1T, 64, 64);
  trans_f2bf<<<(64 * 64 + 255) / 256, blk, 0, stream>>>(Wsl, WslT, 64, 64);
  trans_f2bf<<<(64 * 64 + 255) / 256, blk, 0, stream>>>(Wq, WqT, 64, 64);
  trans_f2bf<<<(64 * 64 + 255) / 256, blk, 0, stream>>>(Wk, WkT, 64, 64);
  trans_f2bf<<<(64 * 64 + 255) / 256, blk, 0, stream>>>(Wv, WvT, 64, 64);
  trans_f2bf<<<(512 * 512 + 255) / 256, blk, 0, stream>>>(Wp, WpT, 512, 512);
  trans_f2bf<<<(512 * 1024 + 255) / 256, blk, 0, stream>>>(W1, W1T, 512, 1024);
  trans_f2bf<<<(1024 * 512 + 255) / 256, blk, 0, stream>>>(W2, W2T, 1024, 512);

  // xm = x @ Wx + bx                              [32768, 512]
  gemm_bf16<bf16, 0><<<dim3(INNERC / 128, MT / 128), blk, 0, stream>>>(
      x_bf, WxT, bx, nullptr, xm_bf, (int)MT, INNERC, DIMC);
  // per-token temp MLP + gumbel softmax -> sw
  token_kernel<<<dim3(NNTOK / 128, BB * HH), blk, 0, stream>>>(
      xm_bf, gumbel, WslT, bsl, tw1T, tb1, tw2, tb2, hbias, sw_bf);
  // dispatch -> st (with snorm normalization)
  dispatch_kernel<<<dim3(BB * HH), blk, 0, stream>>>(sw_bf, xm_bf, st_bf);
  // attention over slice tokens -> out_st^T
  attn_kernel<<<dim3(BB * HH), blk, 0, stream>>>(st_bf, WqT, WkT, WvT, ostT);
  // scatter back -> y
  scatter_kernel<<<dim3(NNTOK / 128, BB * HH), blk, 0, stream>>>(sw_bf, ostT,
                                                                 y_bf);
  // out2 = y @ Wp + bp + x
  gemm_bf16<float, 2><<<dim3(DIMC / 128, MT / 128), blk, 0, stream>>>(
      y_bf, WpT, bp, x, out2, (int)MT, DIMC, INNERC);
  // ln
  ln_kernel<<<dim3((unsigned)MT), dim3(512), 0, stream>>>(out2, ln2g, ln2b,
                                                          ln_bf);
  // h1 = gelu(ln @ W1 + b1)
  gemm_bf16<bf16, 1><<<dim3(HIDC / 128, MT / 128), blk, 0, stream>>>(
      ln_bf, W1T, b1, nullptr, h1_bf, (int)MT, HIDC, DIMC);
  // out = h1 @ W2 + b2 + x
  gemm_bf16<float, 2><<<dim3(DIMC / 128, MT / 128), blk, 0, stream>>>(
      h1_bf, W2T, b2, x, (float*)d_out, (int)MT, DIMC, HIDC);
}
